// ConceptNetEncoder_24343874633941
// MI455X (gfx1250) — matrix operand order repaired and checked by reading, not compiled, not run on hardware
//
#include <hip/hip_runtime.h>
#include <hip/hip_bf16.h>
#include <math.h>

#define V_SZ   100000
#define E_SZ   256
#define NQ     1024       // B*L
#define TOPK   16
#define NSPLIT 32
#define ROWS_PER_SPLIT 3136   // 32*3136 >= 100000
#define NCAND  (NSPLIT*TOPK)  // 512
#define QB     32             // queries per block (2 WMMA column tiles)
#define ROWB   528            // 512B row + 16B TDM pad -> 4-bank skew
#define TILEB  (16*ROWB)      // 8448 B per tile buffer

typedef __bf16 bf16_t;
typedef bf16_t v16bf __attribute__((ext_vector_type(16)));
typedef float  v8f   __attribute__((ext_vector_type(8)));
typedef unsigned int u32x4 __attribute__((ext_vector_type(4)));
typedef int          i32x4 __attribute__((ext_vector_type(4)));
typedef int          i32x8 __attribute__((ext_vector_type(8)));

union ABv { v16bf v; uint4 q[2]; };
union Cv  { v8f  v; float f[8]; };

__device__ __forceinline__ unsigned int f2bf_pack(float a, float b) {
  unsigned ua = __float_as_uint(a), ub = __float_as_uint(b);
  ua = (ua + 0x7FFFu + ((ua >> 16) & 1u)) >> 16;     // RNE
  ub = (ub + 0x7FFFu + ((ub >> 16) & 1u)) >> 16;
  return ua | (ub << 16);
}

// TDM: DMA one 16x256 bf16 tile (rows_remaining clamps OOB rows to zero-fill)
// into LDS at lds_off, with 16B padding after each 512B row (bank skew).
__device__ __forceinline__ void tdm_load_tile(unsigned lds_off,
                                              const unsigned short* gsrc,
                                              int rows_remaining) {
  unsigned long long ga = (unsigned long long)(size_t)gsrc;
  u32x4 g0;
  g0[0] = 1u;                                        // count=1, user mode
  g0[1] = lds_off;                                   // lds_addr (bytes)
  g0[2] = (unsigned)(ga & 0xFFFFFFFFu);              // global_addr[31:0]
  g0[3] = (unsigned)((ga >> 32) & 0x01FFFFFFu)       // global_addr[56:32]
          | (2u << 30);                              // type = 2 (image)
  i32x8 g1;
  g1[0] = (1 << 16)            // data_size = 2 bytes
        | (1 << 20)            // pad_enable
        | (6 << 22)            // pad_interval: 2^6 * 8B = 512B (128 dwords)
        | (3 << 25);           // pad_amount: 4 dwords (16B)
  g1[1] = (256 << 16);                               // tensor_dim0[15:0] = 256
  g1[2] = ((rows_remaining & 0xFFFF) << 16);         // td0[31:16]=0 | td1[15:0]
  g1[3] = ((rows_remaining >> 16) & 0xFFFF)          // td1[31:16]
        | (256 << 16);                               // tile_dim0 = 256
  g1[4] = 16;                                        // tile_dim1=16, tile_dim2=0
  g1[5] = 256;                                       // tensor_dim0_stride[31:0]
  g1[6] = 0;
  g1[7] = 0;
  i32x4 z4 = {0, 0, 0, 0};
  i32x8 z8 = {0, 0, 0, 0, 0, 0, 0, 0};
  __builtin_amdgcn_tensor_load_to_lds(g0, g1, z4, z4, z8, 0);
}

// ---------------- table f32 -> bf16 (packed) ----------------
__global__ void k_convert_table(const float* __restrict__ t, unsigned int* __restrict__ o) {
  size_t i = (size_t)blockIdx.x * blockDim.x + threadIdx.x;  // 8 floats per thread
  size_t base = i * 8;
  if (base >= (size_t)V_SZ * E_SZ) return;
  float4 x0 = *(const float4*)(t + base);
  float4 x1 = *(const float4*)(t + base + 4);
  uint4 r;
  r.x = f2bf_pack(x0.x, x0.y); r.y = f2bf_pack(x0.z, x0.w);
  r.z = f2bf_pack(x1.x, x1.y); r.w = f2bf_pack(x1.z, x1.w);
  *(uint4*)(o + i * 4) = r;
}

// ---------------- gather query embeddings, convert to bf16 ----------------
__global__ void k_gather_emb(const int* __restrict__ ids, const float* __restrict__ t,
                             unsigned int* __restrict__ o) {
  int q = blockIdx.x, lane = threadIdx.x;   // 128 threads: one dword (2 bf16) each
  int id = ids[q];
  const float* src = t + (size_t)id * E_SZ + lane * 2;
  o[(size_t)q * (E_SZ / 2) + lane] = f2bf_pack(src[0], src[1]);
}

// ---------------- pre-pack `a` into WMMA B-operand lane layout ----------------
__global__ void k_pack_aB(const float* __restrict__ a, unsigned int* __restrict__ o) {
  int f = blockIdx.x * blockDim.x + threadIdx.x;   // 32768 dwords
  int d = f & 7, lane = (f >> 3) & 31, c = (f >> 8) & 7, n = f >> 11;
  int j = n * 16 + (lane & 15);
  int k = c * 32 + (lane >> 4) * 16 + d * 2;
  o[f] = f2bf_pack(a[k * E_SZ + j], a[(k + 1) * E_SZ + j]);
}

// ---- fused logits GEMM + running top-16 (TDM double-buffered, 32 queries/block) ----
__global__ __launch_bounds__(32)
void k_logits_topk(const unsigned short* __restrict__ tb16,
                   const unsigned short* __restrict__ emb16,
                   float* __restrict__ cand_val, int* __restrict__ cand_idx) {
  int lane  = threadIdx.x;
  int qbase = blockIdx.x * QB;
  int split = blockIdx.y;
  int rs = split * ROWS_PER_SPLIT;
  int re = min(rs + ROWS_PER_SPLIT, V_SZ);

  __shared__ __align__(16) unsigned char tbuf[2 * TILEB];   // TDM double buffer
  __shared__ float bval[QB][16];
  __shared__ int   bidx[QB][16];
  __shared__ float bmin[QB];
  __shared__ int   bmpos[QB];
  for (int t = 0; t < 16; ++t) {
    int i = lane + 32 * t;
    bval[i >> 4][i & 15] = -3.0e38f;
    bidx[i >> 4][i & 15] = 0;
  }
  bmin[lane] = -3.0e38f;
  bmpos[lane] = 0;

  // B operands: 32 query embeddings, two 16-wide column tiles, in VGPRs for the kernel
  ABv Bm0[8], Bm1[8];
  {
    const unsigned short* eb0 =
        emb16 + (size_t)(qbase + (lane & 15)) * E_SZ + (lane >> 4) * 8;
    const unsigned short* eb1 = eb0 + (size_t)16 * E_SZ;
#pragma unroll
    for (int c = 0; c < 8; ++c) {
      const uint4* p0 = (const uint4*)(eb0 + c * 32);
      const uint4* p1 = (const uint4*)(eb1 + c * 32);
      Bm0[c].q[0] = p0[0]; Bm0[c].q[1] = p0[2];
      Bm1[c].q[0] = p1[0]; Bm1[c].q[1] = p1[2];
    }
  }
  int half = lane >> 4;
  unsigned lds_base = (unsigned)(size_t)(void*)&tbuf[0];

  // prime the pipeline: tile 0 -> buffer 0
  tdm_load_tile(lds_base, tb16 + (size_t)rs * E_SZ, V_SZ - rs);

  int cur = 0;
  for (int tb = rs; tb < re; tb += 16) {
    int nxt = tb + 16;
    if (nxt < re) {  // issue next tile DMA, then wait for current (in-order TENSORcnt)
      tdm_load_tile(lds_base + (unsigned)((cur ^ 1) * TILEB),
                    tb16 + (size_t)nxt * E_SZ, V_SZ - nxt);
      __builtin_amdgcn_s_wait_tensorcnt(1);
    } else {
      __builtin_amdgcn_s_wait_tensorcnt(0);
    }

    // software-pipelined A chunks: load ck+1 while WMMAs consume ck
    const unsigned char* ab = tbuf + cur * TILEB + (lane & 15) * ROWB + half * 16;
    ABv Aa, Ab;
    Aa.q[0] = *(const uint4*)(ab);
    Aa.q[1] = *(const uint4*)(ab + 32);
    Cv c0, c1;
#pragma unroll
    for (int i = 0; i < 8; ++i) { c0.f[i] = 0.f; c1.f[i] = 0.f; }
#pragma unroll
    for (int ck = 0; ck < 8; ++ck) {
      ABv& cu = (ck & 1) ? Ab : Aa;
      ABv& nx = (ck & 1) ? Aa : Ab;
      if (ck < 7) {
        nx.q[0] = *(const uint4*)(ab + (ck + 1) * 64);
        nx.q[1] = *(const uint4*)(ab + (ck + 1) * 64 + 32);
      }
      c0.v = __builtin_amdgcn_wmma_f32_16x16x32_bf16(
          false, cu.v, false, Bm0[ck].v, (short)0, c0.v, false, false);
      c1.v = __builtin_amdgcn_wmma_f32_16x16x32_bf16(
          false, cu.v, false, Bm1[ck].v, (short)0, c1.v, false, false);
    }

    // running top-16 insert: 2 column sets x 2 EXEC-serialized half passes
    for (int cs = 0; cs < 2; ++cs) {
      Cv& cc = cs ? c1 : c0;
      int q = (lane & 15) + cs * 16;
      for (int pass = 0; pass < 2; ++pass) {
        if (half == pass) {
#pragma unroll
          for (int r = 0; r < 8; ++r) {
            int row = tb + r + 8 * half;
            float v = cc.f[r];
            if (row < re && v > bmin[q]) {
              int p = bmpos[q];
              bval[q][p] = v; bidx[q][p] = row;
              float m = bval[q][0]; int mp = 0;
#pragma unroll
              for (int j = 1; j < 16; ++j) { float x = bval[q][j]; if (x < m) { m = x; mp = j; } }
              bmin[q] = m; bmpos[q] = mp;
            }
          }
        }
      }
    }
    cur ^= 1;
  }
  for (int t = 0; t < 16; ++t) {
    int i = lane + 32 * t;
    int qq = i >> 4, j = i & 15;
    size_t o = (size_t)(qbase + qq) * NCAND + (size_t)split * TOPK + j;
    cand_val[o] = bval[qq][j];
    cand_idx[o] = bidx[qq][j];
  }
}

// ---------------- merge 512 candidates -> top-16 per query ----------------
__global__ __launch_bounds__(32)
void k_merge(const float* __restrict__ cand_val, const int* __restrict__ cand_idx,
             int* __restrict__ topidx) {
  int q = blockIdx.x, lane = threadIdx.x;
  float v[16]; int id[16];
#pragma unroll
  for (int j = 0; j < 16; ++j) {
    size_t o = (size_t)q * NCAND + lane + 32 * j;
    v[j] = cand_val[o]; id[j] = cand_idx[o];
  }
  int seli = 0;
  for (int it = 0; it < 16; ++it) {
    float lm = v[0]; int lp = 0;
#pragma unroll
    for (int j = 1; j < 16; ++j) if (v[j] > lm) { lm = v[j]; lp = j; }
    float gv = lm; int gid = id[lp];
#pragma unroll
    for (int off = 16; off > 0; off >>= 1) {
      float ov  = __shfl_xor(gv, off, 32);
      int   oid = __shfl_xor(gid, off, 32);
      if (ov > gv || (ov == gv && oid < gid)) { gv = ov; gid = oid; }
    }
    if (lm == gv && id[lp] == gid) v[lp] = -3.0e38f;  // owner removes winner
    if (lane == it) seli = gid;
  }
  if (lane < 16) topidx[q * 16 + lane] = seli;
}

// ---------------- attention pool over the 16 retrieved concepts ----------------
__global__ __launch_bounds__(32)
void k_attn(const int* __restrict__ topidx, const float* __restrict__ table,
            const unsigned int* __restrict__ aBp, const float* __restrict__ bvec,
            float* __restrict__ out) {
  int q = blockIdx.x, lane = threadIdx.x;
  __shared__ int   sidx[16];
  __shared__ float eacc[16];
  __shared__ float attw[16];
  if (lane < 16) { sidx[lane] = topidx[q * 16 + lane]; eacc[lane] = 0.f; }
  __syncthreads();

  int half = lane >> 4, kq = lane & 15;
  // A operand: h = table[idx] rows, converted f32 -> bf16 into WMMA A layout
  ABv A[8];
#pragma unroll
  for (int c = 0; c < 8; ++c) {
    const float* hp = table + (size_t)sidx[kq] * E_SZ + c * 32 + half * 16;
    float4 x0 = *(const float4*)hp,        x1 = *(const float4*)(hp + 4);
    float4 y0 = *(const float4*)(hp + 8), y1 = *(const float4*)(hp + 12);
    A[c].q[0] = make_uint4(f2bf_pack(x0.x, x0.y), f2bf_pack(x0.z, x0.w),
                           f2bf_pack(x1.x, x1.y), f2bf_pack(x1.z, x1.w));
    A[c].q[1] = make_uint4(f2bf_pack(y0.x, y0.y), f2bf_pack(y0.z, y0.w),
                           f2bf_pack(y1.x, y1.y), f2bf_pack(y1.z, y1.w));
  }
  float ep[8] = {0, 0, 0, 0, 0, 0, 0, 0};
  for (int n = 0; n < 16; ++n) {
    // software-pipelined B chunks: load ck+1 while WMMA consumes ck
    const uint4* bp = (const uint4*)aBp + ((size_t)n * 256 + lane) * 2;
    ABv Ba, Bb;
    Ba.q[0] = bp[0];
    Ba.q[1] = bp[1];
    Cv cc;
#pragma unroll
    for (int i = 0; i < 8; ++i) cc.f[i] = 0.f;
#pragma unroll
    for (int ck = 0; ck < 8; ++ck) {
      ABv& cu = (ck & 1) ? Bb : Ba;
      ABv& nx = (ck & 1) ? Ba : Bb;
      if (ck < 7) {
        nx.q[0] = bp[(ck + 1) * 64];
        nx.q[1] = bp[(ck + 1) * 64 + 1];
      }
      cc.v = __builtin_amdgcn_wmma_f32_16x16x32_bf16(
          false, A[ck].v, false, cu.v, (short)0, cc.v, false, false);
    }
    float bj = bvec[n * 16 + kq];
#pragma unroll
    for (int r = 0; r < 8; ++r) ep[r] += tanhf(cc.f[r]) * bj;   // e[k] partials
  }
#pragma unroll
  for (int r = 0; r < 8; ++r) atomicAdd(&eacc[r + 8 * half], ep[r]);
  __syncthreads();
  if (lane == 0) {
    float mx = eacc[0];
    for (int k = 1; k < 16; ++k) mx = fmaxf(mx, eacc[k]);
    float s = 0.f;
    for (int k = 0; k < 16; ++k) { float e = __expf(eacc[k] - mx); attw[k] = e; s += e; }
    float inv = 1.f / s;
    for (int k = 0; k < 16; ++k) attw[k] *= inv;
  }
  __syncthreads();
#pragma unroll
  for (int t = 0; t < 8; ++t) {
    int e = lane + 32 * t;
    float acc = 0.f;
#pragma unroll
    for (int k = 0; k < 16; ++k) acc += attw[k] * table[(size_t)sidx[k] * E_SZ + e];
    out[(size_t)q * E_SZ + e] = acc;
  }
}

extern "C" void kernel_launch(void* const* d_in, const int* in_sizes, int n_in,
                              void* d_out, int out_size, void* d_ws, size_t ws_size,
                              hipStream_t stream) {
  const int*   ids   = (const int*)d_in[0];
  const float* table = (const float*)d_in[1];
  const float* amat  = (const float*)d_in[2];
  const float* bvec  = (const float*)d_in[3];
  (void)in_sizes; (void)n_in; (void)out_size; (void)ws_size;

  char* ws = (char*)d_ws;
  size_t off = 0;
  auto carve = [&](size_t bytes) -> char* {
    char* p = ws + off;
    off += (bytes + 255) & ~(size_t)255;
    return p;
  };
  unsigned short* tb16     = (unsigned short*)carve((size_t)V_SZ * E_SZ * 2);
  unsigned short* emb16    = (unsigned short*)carve((size_t)NQ * E_SZ * 2);
  unsigned int*   aBp      = (unsigned int*)carve((size_t)16 * 8 * 32 * 8 * 4);
  float*          cand_val = (float*)carve((size_t)NQ * NCAND * 4);
  int*            cand_idx = (int*)carve((size_t)NQ * NCAND * 4);
  int*            topidx   = (int*)carve((size_t)NQ * TOPK * 4);

  hipLaunchKernelGGL(k_convert_table, dim3((V_SZ * E_SZ / 8 + 255) / 256), dim3(256), 0, stream,
                     table, (unsigned int*)tb16);
  hipLaunchKernelGGL(k_gather_emb, dim3(NQ), dim3(128), 0, stream,
                     ids, table, (unsigned int*)emb16);
  hipLaunchKernelGGL(k_pack_aB, dim3(128), dim3(256), 0, stream, amat, aBp);
  hipLaunchKernelGGL(k_logits_topk, dim3(NQ / QB, NSPLIT), dim3(32), 0, stream,
                     tb16, emb16, cand_val, cand_idx);
  hipLaunchKernelGGL(k_merge, dim3(NQ), dim3(32), 0, stream, cand_val, cand_idx, topidx);
  hipLaunchKernelGGL(k_attn, dim3(NQ), dim3(32), 0, stream, topidx, table, aBp, bvec,
                     (float*)d_out);
}
